// PointTransformerBlock_35098472743106
// MI455X (gfx1250) — matrix operand order repaired and checked by reading, not compiled, not run on hardware
//
#include <hip/hip_runtime.h>
#include <stdint.h>

// ---------------------------------------------------------------------------
// Point Transformer block for MI455X (gfx1250, wave32, WMMA 16x16x32 f16)
// B=4, N=16384, K=16, D=128, H=256
// ---------------------------------------------------------------------------

#define Bb 4
#define Nn 16384
#define Kk 16
#define Dd 128
#define Hh 256
#define TILE 16              // query points per workgroup
#define ROWS (TILE * Kk)     // 256 neighbor rows per workgroup
#define NB16 (Nn / TILE)     // 1024 tiles per batch

typedef __attribute__((ext_vector_type(16))) _Float16 v16h;
typedef __attribute__((ext_vector_type(8)))  float    v8f;

union FragU { v16h v; uint4 q[2]; };

// ---- WMMA wrapper ----------------------------------------------------------
__device__ __forceinline__ v8f wmma16(v16h a, v16h b, v8f c) {
  return __builtin_amdgcn_wmma_f32_16x16x32_f16(
      /*neg_a=*/false, a, /*neg_b=*/false, b,
      /*c_mod=*/(short)0, c, /*reuse_a=*/false, /*reuse_b=*/false);
}

// A fragment (16x32, 16-bit): lane m = lane&15; k = kbase+[0..7] and +[16..23]
__device__ __forceinline__ v16h load_a(const _Float16* s, int ld, int rowBase,
                                       int k0, int lane) {
  int m  = rowBase + (lane & 15);
  int kb = k0 + ((lane >> 4) << 3);
  const _Float16* p = s + m * ld + kb;
  FragU u;
  u.q[0] = *(const uint4*)(p);
  u.q[1] = *(const uint4*)(p + 16);
  return u.v;
}

// B fragment (32x16, 16-bit) from pre-transposed weight WT (Nw x Kw row-major):
// lane n = lane&15; k = k0 + (lane>=16 ? 8 : 0) + [0..7] and +[16..23]
__device__ __forceinline__ v16h load_b(const _Float16* wt, int ldk, int n0,
                                       int k0, int lane) {
  int n  = n0 + (lane & 15);
  int kb = k0 + ((lane >> 4) << 3);
  const _Float16* p = wt + n * ldk + kb;
  FragU u;
  u.q[0] = *(const uint4*)(p);
  u.q[1] = *(const uint4*)(p + 16);
  return u.v;
}

// store a 16x16 f32 accumulator tile into LDS as f16 (optionally relu)
__device__ __forceinline__ void store_tile_h(_Float16* d, int ld, int rowBase,
                                             int n0, v8f acc, bool relu,
                                             int lane) {
  int n  = n0 + (lane & 15);
  int mh = (lane >> 4) << 3;
  #pragma unroll
  for (int i = 0; i < 8; ++i) {
    float v = acc[i];
    if (relu) v = fmaxf(v, 0.0f);
    d[(rowBase + mh + i) * ld + n] = (_Float16)v;
  }
}

__device__ __forceinline__ float wave_sum32(float x) {
  #pragma unroll
  for (int off = 16; off; off >>= 1) x += __shfl_xor(x, off, 32);
  return x;
}

// CDNA5 async global->LDS copy (16 bytes). ASYNCcnt tracked.
#define USE_ASYNC_GATHER 1
__device__ __forceinline__ void async_ld16(uint32_t lds_off, const void* g) {
#if USE_ASYNC_GATHER
  asm volatile("global_load_async_to_lds_b128 %0, %1, off"
               :: "v"(lds_off), "v"((unsigned long long)(uintptr_t)g)
               : "memory");
#else
  (void)lds_off; (void)g;
#endif
}

// ---------------------------------------------------------------------------
// Kernel 1: transpose weight (Kw x Nw f32) -> (Nw x Kw f16)
// ---------------------------------------------------------------------------
__global__ void k_transpose_h(const float* __restrict__ src,
                              _Float16* __restrict__ dst, int Kw, int Nw) {
  int i = blockIdx.x * 256 + threadIdx.x;
  if (i < Kw * Nw) {
    int n = i / Kw, k = i % Kw;
    dst[i] = (_Float16)src[k * Nw + n];
  }
}

// ---------------------------------------------------------------------------
// Kernel 2: LN1 over feats -> x (f16).  One wave per point (D=128, 4/lane).
// ---------------------------------------------------------------------------
__global__ void k_ln1(const float* __restrict__ feats,
                      const float* __restrict__ g, const float* __restrict__ b,
                      _Float16* __restrict__ xo, int npts) {
  int wave = (blockIdx.x * blockDim.x + threadIdx.x) >> 5;
  int lane = threadIdx.x & 31;
  if (wave >= npts) return;
  const float* row = feats + (size_t)wave * Dd;
  float4 v = *(const float4*)(row + lane * 4);
  float s  = v.x + v.y + v.z + v.w;
  float s2 = v.x * v.x + v.y * v.y + v.z * v.z + v.w * v.w;
  s = wave_sum32(s); s2 = wave_sum32(s2);
  float mean = s * (1.0f / Dd);
  float inv  = rsqrtf(s2 * (1.0f / Dd) - mean * mean + 1e-5f);
  float r[4] = {v.x, v.y, v.z, v.w};
  _Float16* o = xo + (size_t)wave * Dd + lane * 4;
  #pragma unroll
  for (int i = 0; i < 4; ++i) {
    int d = lane * 4 + i;
    o[i] = (_Float16)((r[i] - mean) * inv * g[d] + b[d]);
  }
}

// ---------------------------------------------------------------------------
// Kernel 3: fused attention block for one 16-point tile (256 thr = 8 waves)
// ---------------------------------------------------------------------------
__global__ void k_attn(const float* __restrict__ xyz,
                       const float* __restrict__ feats,
                       const int* __restrict__ idx,
                       const _Float16* __restrict__ xh,
                       const float* __restrict__ Wpos1,
                       const _Float16* __restrict__ WqT,
                       const _Float16* __restrict__ WkvT,
                       const _Float16* __restrict__ Wpos2T,
                       const _Float16* __restrict__ Wattn1T,
                       const _Float16* __restrict__ Wattn2T,
                       const _Float16* __restrict__ WprojT,
                       float* __restrict__ out) {
  extern __shared__ __align__(16) char smem[];
  _Float16* nb = (_Float16*)smem;        // 256x128 : neigh x   -> later a1
  _Float16* kb = nb + ROWS * Dd;         // 256x128 : k         -> later s
  _Float16* vb = kb + ROWS * Dd;         // 256x128 : v
  _Float16* pb = vb + ROWS * Dd;         // 256x128 : pos
  _Float16* xq = pb + ROWS * Dd;         // 16x128  : x tile    -> later out f16
  _Float16* qb = xq + TILE * Dd;         // 16x128  : q
  float* rel   = (float*)(qb + TILE * Dd); // 256x4 : rel xyz
  float* outf  = rel + ROWS * 4;         // 16x128  : attention output f32
  float* w1s   = outf + TILE * Dd;       // 3x128   : Wpos1
  int*   idxs  = (int*)(w1s + 3 * Dd);   // 256

  const int t    = threadIdx.x;
  const int lane = t & 31;
  const int w    = t >> 5;               // wave id = column tile
  const int b    = blockIdx.x / NB16;
  const int p0   = (blockIdx.x % NB16) * TILE;
  const size_t base = (size_t)b * Nn;

  // ---------------- Phase 0: indices, rel, own rows, neighbor gather -------
  idxs[t] = idx[(base + p0 + (t >> 4)) * Kk + (t & 15)];
  if (t < 3 * Dd) w1s[t] = Wpos1[t];
  __syncthreads();
  {
    int p = t >> 4;
    int nbr = idxs[t];
    #pragma unroll
    for (int c = 0; c < 3; ++c)
      rel[t * 4 + c] = xyz[(base + nbr) * 3 + c] - xyz[(base + p0 + p) * 3 + c];
    // own x rows (16 rows x 16 chunks of 16B)
    int row = t >> 4, ch = t & 15;
    *(uint4*)(xq + row * Dd + ch * 8) =
        *(const uint4*)(xh + (base + p0 + row) * Dd + ch * 8);
    // neighbor rows: 16 iterations x (16 rows x 16 chunks) async to LDS
    for (int it = 0; it < 16; ++it) {
      int r = it * 16 + (t >> 4);
      const _Float16* src = xh + (base + idxs[r]) * Dd + ch * 8;
      uint32_t dst = (uint32_t)(uintptr_t)(nb + r * Dd + ch * 8);
#if USE_ASYNC_GATHER
      async_ld16(dst, src);
#else
      *(uint4*)(nb + r * Dd + ch * 8) = *(const uint4*)src;
#endif
    }
  }
#if USE_ASYNC_GATHER
  asm volatile("s_wait_asynccnt 0x0" ::: "memory");
#endif
  __syncthreads();

  // ---------------- Phase 1: q = x @ Wq  (16x128) ---------------------------
  {
    v8f acc = {};
    #pragma unroll
    for (int kk = 0; kk < 4; ++kk) {
      v16h a  = load_a(xq, Dd, 0, kk * 32, lane);
      v16h bb = load_b(WqT, Dd, w * 16, kk * 32, lane);
      acc = wmma16(a, bb, acc);
    }
    store_tile_h(qb, Dd, 0, w * 16, acc, false, lane);
  }
  __syncthreads();

  // ---------------- Phase 2: k,v = neigh_x @ Wkv (256x256) ------------------
  {
    v16h bk[4], bv[4];
    #pragma unroll
    for (int kk = 0; kk < 4; ++kk) {
      bk[kk] = load_b(WkvT, Dd, w * 16,       kk * 32, lane);
      bv[kk] = load_b(WkvT, Dd, Dd + w * 16,  kk * 32, lane);
    }
    for (int rt = 0; rt < TILE; ++rt) {
      v8f ak = {}, av = {};
      #pragma unroll
      for (int kk = 0; kk < 4; ++kk) {
        v16h a = load_a(nb, Dd, rt * 16, kk * 32, lane);
        ak = wmma16(a, bk[kk], ak);
        av = wmma16(a, bv[kk], av);
      }
      store_tile_h(kb, Dd, rt * 16, w * 16, ak, false, lane);
      store_tile_h(vb, Dd, rt * 16, w * 16, av, false, lane);
    }
  }
  __syncthreads();

  // ---------------- Phase 3: pos = relu(rel@Wpos1) @ Wpos2 ------------------
  {
    v16h bf[4];
    #pragma unroll
    for (int kk = 0; kk < 4; ++kk)
      bf[kk] = load_b(Wpos2T, Dd, w * 16, kk * 32, lane);
    for (int rt = 0; rt < TILE; ++rt) {
      v8f acc = {};
      int m = rt * 16 + (lane & 15);
      float r0 = rel[m * 4], r1 = rel[m * 4 + 1], r2 = rel[m * 4 + 2];
      #pragma unroll
      for (int kk = 0; kk < 4; ++kk) {
        int kbs = kk * 32 + ((lane >> 4) << 3);
        v16h a;
        #pragma unroll
        for (int i = 0; i < 16; ++i) {
          int k = kbs + (i < 8 ? i : i + 8);   // [kbs..kbs+7], [kbs+16..kbs+23]
          float p1 = fmaxf(r0 * w1s[k] + r1 * w1s[Dd + k] + r2 * w1s[2 * Dd + k], 0.0f);
          a[i] = (_Float16)p1;
        }
        acc = wmma16(a, bf[kk], acc);
      }
      store_tile_h(pb, Dd, rt * 16, w * 16, acc, false, lane);
    }
  }
  __syncthreads();

  // ---------------- Phase 4: s = q - k + pos  (into kb) ---------------------
  {
    int p = t >> 4;
    _Float16* srow = kb + t * Dd;
    const _Float16* prow = pb + t * Dd;
    const _Float16* qrow = qb + p * Dd;
    for (int d = 0; d < Dd; ++d)
      srow[d] = (_Float16)((float)qrow[d] - (float)srow[d] + (float)prow[d]);
  }
  __syncthreads();

  // ---------------- Phase 5: a1 = relu(s @ Wattn1)  (into nb) ---------------
  {
    v16h bf[4];
    #pragma unroll
    for (int kk = 0; kk < 4; ++kk)
      bf[kk] = load_b(Wattn1T, Dd, w * 16, kk * 32, lane);
    for (int rt = 0; rt < TILE; ++rt) {
      v8f acc = {};
      #pragma unroll
      for (int kk = 0; kk < 4; ++kk)
        acc = wmma16(load_a(kb, Dd, rt * 16, kk * 32, lane), bf[kk], acc);
      store_tile_h(nb, Dd, rt * 16, w * 16, acc, true, lane);
    }
  }
  __syncthreads();

  // ------- Phase 6: attn = a1 @ Wattn2; softmax over K; out = attn*(v+pos) --
  {
    v16h bf[4];
    #pragma unroll
    for (int kk = 0; kk < 4; ++kk)
      bf[kk] = load_b(Wattn2T, Dd, w * 16, kk * 32, lane);
    int n  = w * 16 + (lane & 15);
    int mh = (lane >> 4) << 3;
    for (int p = 0; p < TILE; ++p) {
      v8f acc = {};
      #pragma unroll
      for (int kk = 0; kk < 4; ++kk)
        acc = wmma16(load_a(nb, Dd, p * 16, kk * 32, lane), bf[kk], acc);
      // softmax over the 16 rows (= the K neighbors of point p) of this tile
      float mx = acc[0];
      #pragma unroll
      for (int i = 1; i < 8; ++i) mx = fmaxf(mx, acc[i]);
      mx = fmaxf(mx, __shfl_xor(mx, 16, 32));
      float e[8], sum = 0.0f;
      #pragma unroll
      for (int i = 0; i < 8; ++i) { e[i] = __expf(acc[i] - mx); sum += e[i]; }
      sum += __shfl_xor(sum, 16, 32);
      float inv = 1.0f / sum;
      float contrib = 0.0f;
      #pragma unroll
      for (int i = 0; i < 8; ++i) {
        int r = p * 16 + mh + i;
        float vp = (float)vb[r * Dd + n] + (float)pb[r * Dd + n];
        contrib += e[i] * inv * vp;
      }
      contrib += __shfl_xor(contrib, 16, 32);
      if (lane < 16) outf[p * Dd + n] = contrib;
    }
  }
  __syncthreads();

  // ---------------- Phase 7: out -> f16 (reuse xq) --------------------------
  for (int i = t; i < TILE * Dd; i += 256) xq[i] = (_Float16)outf[i];
  __syncthreads();

  // ---------------- Phase 8: d_out = feats + out @ Wproj --------------------
  {
    v8f acc = {};
    #pragma unroll
    for (int kk = 0; kk < 4; ++kk) {
      v16h a  = load_a(xq, Dd, 0, kk * 32, lane);
      v16h bb = load_b(WprojT, Dd, w * 16, kk * 32, lane);
      acc = wmma16(a, bb, acc);
    }
    int n  = w * 16 + (lane & 15);
    int mh = (lane >> 4) << 3;
    #pragma unroll
    for (int i = 0; i < 8; ++i) {
      size_t g = (base + p0 + mh + i) * Dd + n;
      out[g] = feats[g] + acc[i];
    }
  }
}

// ---------------------------------------------------------------------------
// Kernel 4: fused LN2 + FFN, in-place on d_out, one 16-point tile per block
// ---------------------------------------------------------------------------
__global__ void k_ffn(float* __restrict__ out,
                      const float* __restrict__ g, const float* __restrict__ b,
                      const _Float16* __restrict__ Wffn1T,
                      const _Float16* __restrict__ Wffn2T) {
  __shared__ float    f1s[TILE * Dd];
  __shared__ _Float16 yb[TILE * Dd];
  __shared__ _Float16 hb[TILE * Hh];

  const int t    = threadIdx.x;
  const int lane = t & 31;
  const int w    = t >> 5;
  const size_t rowBase = (size_t)blockIdx.x * TILE * Dd;

  // load tile of feats1
  #pragma unroll
  for (int i = 0; i < 2; ++i)
    *(float4*)(f1s + t * 8 + i * 4) = *(const float4*)(out + rowBase + t * 8 + i * 4);
  __syncthreads();

  // LN2: each wave handles 2 points
  #pragma unroll
  for (int pp = 0; pp < 2; ++pp) {
    int pt = 2 * w + pp;
    float4 v = *(float4*)(f1s + pt * Dd + lane * 4);
    float s  = wave_sum32(v.x + v.y + v.z + v.w);
    float s2 = wave_sum32(v.x * v.x + v.y * v.y + v.z * v.z + v.w * v.w);
    float mean = s * (1.0f / Dd);
    float inv  = rsqrtf(s2 * (1.0f / Dd) - mean * mean + 1e-5f);
    float r[4] = {v.x, v.y, v.z, v.w};
    #pragma unroll
    for (int i = 0; i < 4; ++i) {
      int d = lane * 4 + i;
      yb[pt * Dd + d] = (_Float16)((r[i] - mean) * inv * g[d] + b[d]);
    }
  }
  __syncthreads();

  // h = relu(y @ Wffn1): 16 col tiles, wave does w and w+8
  #pragma unroll
  for (int c = 0; c < 2; ++c) {
    int n0 = (w + c * 8) * 16;
    v8f acc = {};
    #pragma unroll
    for (int kk = 0; kk < 4; ++kk)
      acc = wmma16(load_a(yb, Dd, 0, kk * 32, lane),
                   load_b(Wffn1T, Dd, n0, kk * 32, lane), acc);
    store_tile_h(hb, Hh, 0, n0, acc, true, lane);
  }
  __syncthreads();

  // o = h @ Wffn2 (K=256); final = f1 + o
  {
    v8f acc = {};
    #pragma unroll
    for (int kk = 0; kk < 8; ++kk)
      acc = wmma16(load_a(hb, Hh, 0, kk * 32, lane),
                   load_b(Wffn2T, Hh, w * 16, kk * 32, lane), acc);
    int n  = w * 16 + (lane & 15);
    int mh = (lane >> 4) << 3;
    #pragma unroll
    for (int i = 0; i < 8; ++i) {
      int p = mh + i;
      out[rowBase + p * Dd + n] = f1s[p * Dd + n] + acc[i];
    }
  }
}

// ---------------------------------------------------------------------------
// Host launcher
// ---------------------------------------------------------------------------
extern "C" void kernel_launch(void* const* d_in, const int* in_sizes, int n_in,
                              void* d_out, int out_size, void* d_ws, size_t ws_size,
                              hipStream_t stream) {
  const float* xyz    = (const float*)d_in[0];
  const float* feats  = (const float*)d_in[1];
  const int*   idx    = (const int*)  d_in[2];
  const float* ln1_g  = (const float*)d_in[3];
  const float* ln1_b  = (const float*)d_in[4];
  const float* ln2_g  = (const float*)d_in[5];
  const float* ln2_b  = (const float*)d_in[6];
  const float* Wq     = (const float*)d_in[7];
  const float* Wkv    = (const float*)d_in[8];
  const float* Wpos1  = (const float*)d_in[9];
  const float* Wpos2  = (const float*)d_in[10];
  const float* Wattn1 = (const float*)d_in[11];
  const float* Wattn2 = (const float*)d_in[12];
  const float* Wproj  = (const float*)d_in[13];
  const float* Wffn1  = (const float*)d_in[14];
  const float* Wffn2  = (const float*)d_in[15];
  float* out = (float*)d_out;

  _Float16* h = (_Float16*)d_ws;
  _Float16* WqT     = h;                  // 128*128
  _Float16* WkvT    = WqT     + 128 * 128; // 256*128
  _Float16* Wpos2T  = WkvT    + 256 * 128;
  _Float16* Wattn1T = Wpos2T  + 128 * 128;
  _Float16* Wattn2T = Wattn1T + 128 * 128;
  _Float16* WprojT  = Wattn2T + 128 * 128;
  _Float16* Wffn1T  = WprojT  + 128 * 128; // 256*128
  _Float16* Wffn2T  = Wffn1T  + 256 * 128; // 128*256
  _Float16* xh      = Wffn2T  + 128 * 256; // B*N*D f16

  auto tr = [&](const float* src, _Float16* dst, int Kw, int Nw) {
    int tot = Kw * Nw;
    k_transpose_h<<<(tot + 255) / 256, 256, 0, stream>>>(src, dst, Kw, Nw);
  };
  tr(Wq,     WqT,     128, 128);
  tr(Wkv,    WkvT,    128, 256);
  tr(Wpos2,  Wpos2T,  128, 128);
  tr(Wattn1, Wattn1T, 128, 128);
  tr(Wattn2, Wattn2T, 128, 128);
  tr(Wproj,  WprojT,  128, 128);
  tr(Wffn1,  Wffn1T,  128, 256);
  tr(Wffn2,  Wffn2T,  256, 128);

  int npts = Bb * Nn;
  k_ln1<<<npts / 8, 256, 0, stream>>>(feats, ln1_g, ln1_b, xh, npts);

  size_t smem = (size_t)(4 * ROWS * Dd + 2 * TILE * Dd) * sizeof(_Float16)
              + (size_t)(ROWS * 4 + TILE * Dd + 3 * Dd) * sizeof(float)
              + (size_t)ROWS * sizeof(int);   // = 285,184 bytes (< 320 KB WGP LDS)
  k_attn<<<Bb * NB16, 256, smem, stream>>>(xyz, feats, idx, xh, Wpos1,
                                           WqT, WkvT, Wpos2T, Wattn1T,
                                           Wattn2T, WprojT, out);

  k_ffn<<<Bb * NB16, 256, 0, stream>>>(out, ln2_g, ln2_b, Wffn1T, Wffn2T);
}
